// LSTMModel_1460288881069
// MI455X (gfx1250) — matrix operand order repaired and checked by reading, not compiled, not run on hardware
//
#include <hip/hip_runtime.h>

// ---------------------------------------------------------------------------
// Fused 2-layer LSTM (H=64) + FC for gfx1250.
// Recurrent/input GEMMs: split-bf16 (hi+lo) v_wmma_f32_16x16x32_bf16
//   h@W ~= hi@Whi + hi@Wlo + lo@Whi  (error ~2^-17: fp32-grade accuracy)
// x-projection (K=6->8): exact fp32 v_wmma_f32_16x16x4_f32.
// 32 batch rows / WG (128 WGs), both layers' state per-WG, 2 barriers/step.
// ---------------------------------------------------------------------------

typedef __attribute__((ext_vector_type(2)))  float  f32x2;
typedef __attribute__((ext_vector_type(8)))  float  f32x8;
typedef __attribute__((ext_vector_type(8)))  __bf16 bf16x8;
typedef __attribute__((ext_vector_type(16))) __bf16 bf16x16;

#define T_LEN   512
#define IN_F    6
#define H       64
#define OUT_F   3
#define MTILE   32
#define THREADS 256
#define TC      16
#define XS_RS   (TC * 8 + 4)     // 132 floats, bank-spread x rows
#define HB_RS   72               // bf16 h row stride (bank-spread)
#define HB_PLANE (MTILE * HB_RS) // 2304 bf16 per [layer][buf][hi/lo] plane

// ---- activations: prefer native v_tanh_f32 -------------------------------
#if __has_builtin(__builtin_amdgcn_tanhf)
__device__ __forceinline__ float fast_tanh(float x) { return __builtin_amdgcn_tanhf(x); }
#elif __has_builtin(__builtin_amdgcn_tanh_f32)
__device__ __forceinline__ float fast_tanh(float x) { return __builtin_amdgcn_tanh_f32(x); }
#else
__device__ __forceinline__ float fast_tanh(float x) {
  return 2.0f * __builtin_amdgcn_rcpf(1.0f + __builtin_amdgcn_exp2f(x * -2.885390081777926f)) - 1.0f;
}
#endif
__device__ __forceinline__ float fast_sigmoid(float x) {
  return 0.5f + 0.5f * fast_tanh(0.5f * x);
}

// ---- WMMA wrappers --------------------------------------------------------
__device__ __forceinline__ f32x8 wmma_bf16(bf16x16 a, bf16x16 b, f32x8 c) {
  return __builtin_amdgcn_wmma_f32_16x16x32_bf16(false, a, false, b, (short)0, c, false, false);
}
__device__ __forceinline__ f32x8 wmma_f32(f32x2 a, f32x2 b, f32x8 c) {
  return __builtin_amdgcn_wmma_f32_16x16x4_f32(false, a, false, b, (short)0, c, false, false);
}

// B fragment: 16 contiguous bf16 per lane (K chunks [c,c+8),[c+16,c+24) at column n) -> 2x b128
__device__ __forceinline__ bf16x16 load_b_frag(const __bf16* p) {
  bf16x8 a = *(const bf16x8*)p;
  bf16x8 b = *(const bf16x8*)(p + 8);
  return __builtin_shufflevector(a, b, 0,1,2,3,4,5,6,7,8,9,10,11,12,13,14,15);
}
// A fragment: per ISA 16-bit A layout, lane needs K chunks [c,c+8) and [c+16,c+24)
__device__ __forceinline__ bf16x16 load_a_frag(const __bf16* p) {
  bf16x8 a = *(const bf16x8*)p;
  bf16x8 b = *(const bf16x8*)(p + 16);
  return __builtin_shufflevector(a, b, 0,1,2,3,4,5,6,7,8,9,10,11,12,13,14,15);
}

// K=64 GEMM as split-bf16: 6 x wmma_bf16 per gate tile, 4-way acc interleave.
// bstream layout: [tile(4)][slot(4): hi_ks0,hi_ks1,lo_ks0,lo_ks1][lane(32)][16]
__device__ __forceinline__ void gemm_k64_split(const __bf16* __restrict__ bstream,
                                               const __bf16* __restrict__ ahiP,
                                               const __bf16* __restrict__ aloP,
                                               int lane, f32x8 acc[4]) {
  bf16x16 Ah[2], Al[2];
#pragma unroll
  for (int ks = 0; ks < 2; ++ks) {
    Ah[ks] = load_a_frag(ahiP + ks * 32);
    Al[ks] = load_a_frag(aloP + ks * 32);
  }
#pragma unroll
  for (int ks = 0; ks < 2; ++ks) {
    bf16x16 Bh[4], Bl[4];
#pragma unroll
    for (int g = 0; g < 4; ++g) {
      Bh[g] = load_b_frag(bstream + ((g * 4 + ks) * 32 + lane) * 16);
      Bl[g] = load_b_frag(bstream + ((g * 4 + 2 + ks) * 32 + lane) * 16);
    }
#pragma unroll
    for (int g = 0; g < 4; ++g) acc[g] = wmma_bf16(Ah[ks], Bh[g], acc[g]);
#pragma unroll
    for (int g = 0; g < 4; ++g) acc[g] = wmma_bf16(Ah[ks], Bl[g], acc[g]);
#pragma unroll
    for (int g = 0; g < 4; ++g) acc[g] = wmma_bf16(Al[ks], Bh[g], acc[g]);
  }
}

// LSTM cell elementwise; c stays in VGPRs, h written as bf16 hi/lo planes.
__device__ __forceinline__ void cell_update(const f32x8 acc[4], float c[8],
                                            __bf16* __restrict__ hhi,
                                            __bf16* __restrict__ hlo,
                                            int mt, int lane, int wv4) {
  const int mbase = mt * 16 + ((lane < 16) ? 0 : 8);
  const int jcol  = wv4 * 16 + (lane & 15);
#pragma unroll
  for (int v = 0; v < 8; ++v) {
    float iv = fast_sigmoid(acc[0][v]);
    float fv = fast_sigmoid(acc[1][v]);
    float gv = fast_tanh(acc[2][v]);
    float ov = fast_sigmoid(acc[3][v]);
    c[v] = fv * c[v] + iv * gv;
    float hv = ov * fast_tanh(c[v]);
    __bf16 hb = (__bf16)hv;
    hhi[(mbase + v) * HB_RS + jcol] = hb;
    hlo[(mbase + v) * HB_RS + jcol] = (__bf16)(hv - (float)hb);
  }
}

extern "C" __global__ __launch_bounds__(THREADS, 1)
void lstm2_fused_wmma(const float* __restrict__ x,
                      const float* __restrict__ Wih0, const float* __restrict__ Whh0,
                      const float* __restrict__ bih0, const float* __restrict__ bhh0,
                      const float* __restrict__ Wih1, const float* __restrict__ Whh1,
                      const float* __restrict__ bih1, const float* __restrict__ bhh1,
                      const float* __restrict__ Wfc,  const float* __restrict__ bfc,
                      float* __restrict__ out)
{
  __shared__ __bf16 lds_w[3 * 32768];              // 192KB: Whh0/Wih1/Whh1 hi+lo frags
  __shared__ float  lds_wx[2048];                  // 8KB: Wih0 f32 frags (K padded 6->8)
  __shared__ float  lds_xs[MTILE * XS_RS];         // x chunk (f32, padded)
  __shared__ __bf16 lds_hb[2 * 2 * 2 * HB_PLANE];  // [layer][pingpong][hi/lo][row][k]

  const int tid  = threadIdx.x;
  const int lane = tid & 31;
  const int wave = tid >> 5;
  const int wv4  = wave & 3;
  const int mt   = wave >> 2;
  const int b0   = blockIdx.x * MTILE;
  const int l15  = lane & 15;
  const int klo  = (lane < 16) ? 0 : 2;   // f32 fragment K offset
  const int aoff = (lane < 16) ? 0 : 8;   // bf16 A fragment K offset

  // ---- one-time staging: split weights into bf16 hi/lo fragment streams ----
  {
    const float* Ws[3] = { Whh0, Wih1, Whh1 };     // all [256][64] row-major
#pragma unroll 1
    for (int q = 0; q < 3; ++q) {
      const float* W = Ws[q];
      __bf16* dst = &lds_w[q * 32768];
      for (int f = tid; f < 2048; f += THREADS) {
        int fl = f & 31, slot = (f >> 5) & 3, ftile = (f >> 7) & 3, fwv = f >> 9;
        int n  = ftile * 64 + fwv * 16 + (fl & 15);
        int c1 = (slot & 1) * 32 + ((fl < 16) ? 0 : 8);
        bool lo = slot >= 2;
        __bf16* d = dst + f * 16;
#pragma unroll
        for (int e = 0; e < 16; ++e) {
          float w = W[n * 64 + c1 + e + ((e >= 8) ? 8 : 0)];
          __bf16 whi = (__bf16)w;
          d[e] = lo ? (__bf16)(w - (float)whi) : whi;
        }
      }
    }
    // Wih0: [256][6] fp32, K padded to 8 -> 2 f32 k-steps
    for (int f = tid; f < 1024; f += THREADS) {
      int fl = f & 31, fkk = (f >> 5) & 1, ftile = (f >> 6) & 3, fwv = f >> 8;
      int n  = ftile * 64 + fwv * 16 + (fl & 15);
      int k0 = fkk * 4 + ((fl < 16) ? 0 : 2);
      f32x2 v;
      v.x = (k0     < IN_F) ? Wih0[n * IN_F + k0]     : 0.0f;
      v.y = (k0 + 1 < IN_F) ? Wih0[n * IN_F + k0 + 1] : 0.0f;
      *(f32x2*)(lds_wx + f * 2) = v;
    }
    for (int i = tid; i < MTILE * XS_RS; i += THREADS) lds_xs[i] = 0.0f;
    for (int i = tid; i < 2 * 2 * 2 * HB_PLANE; i += THREADS) lds_hb[i] = (__bf16)0.0f;
  }

  // ---- per-lane fused biases ----
  float bias0[4], bias1[4];
  {
    int nc = wv4 * 16 + l15;
#pragma unroll
    for (int g = 0; g < 4; ++g) {
      bias0[g] = bih0[g * 64 + nc] + bhh0[g * 64 + nc];
      bias1[g] = bih1[g * 64 + nc] + bhh1[g * 64 + nc];
    }
  }
  float c0[8], c1[8];
#pragma unroll
  for (int v = 0; v < 8; ++v) { c0[v] = 0.0f; c1[v] = 0.0f; }

  const int arow = mt * 16 + l15;
  __syncthreads();

  int pb = 0;
#pragma unroll 1
  for (int t = 0; t < T_LEN; ++t) {
    const int tc = t & (TC - 1);
    if (tc == 0) {
      for (int idx = tid; idx < MTILE * TC * IN_F; idx += THREADS) {
        int r  = idx / (TC * IN_F);
        int pp = idx - r * (TC * IN_F);
        int tt = pp / IN_F, ff = pp - tt * IN_F;
        lds_xs[r * XS_RS + tt * 8 + ff] =
            x[(size_t)(b0 + r) * (T_LEN * IN_F) + (size_t)(t + tt) * IN_F + ff];
      }
      if (tid < MTILE && t + TC < T_LEN)  // global_prefetch_b8 of next chunk
        __builtin_prefetch(&x[(size_t)(b0 + tid) * (T_LEN * IN_F) +
                              (size_t)(t + TC) * IN_F], 0, 1);
      __syncthreads();
    }

    // ===================== layer 0 =====================
    f32x8 acc[4];
#pragma unroll
    for (int g = 0; g < 4; ++g) {
      f32x8 a;
#pragma unroll
      for (int v = 0; v < 8; ++v) a[v] = bias0[g];
      acc[g] = a;
    }
    // input projection x_t @ Wih0^T: exact fp32 wmma (K=8)
    {
      const float* xptr = &lds_xs[arow * XS_RS + tc * 8 + klo];
#pragma unroll
      for (int kk = 0; kk < 2; ++kk) {
        f32x2 a = *(const f32x2*)(xptr + kk * 4);
#pragma unroll
        for (int g = 0; g < 4; ++g) {
          f32x2 b = *(const f32x2*)(lds_wx + (((wv4 * 4 + g) * 2 + kk) * 32 + lane) * 2);
          acc[g] = wmma_f32(a, b, acc[g]);
        }
      }
    }
    // recurrent h0_{t-1} @ Whh0^T (split-bf16)
    gemm_k64_split(&lds_w[0 * 32768 + wv4 * 8192],
                   &lds_hb[((0 * 2 + pb) * 2 + 0) * HB_PLANE + arow * HB_RS + aoff],
                   &lds_hb[((0 * 2 + pb) * 2 + 1) * HB_PLANE + arow * HB_RS + aoff],
                   lane, acc);
    cell_update(acc, c0,
                &lds_hb[((0 * 2 + (1 - pb)) * 2 + 0) * HB_PLANE],
                &lds_hb[((0 * 2 + (1 - pb)) * 2 + 1) * HB_PLANE], mt, lane, wv4);
    __syncthreads();

    // ===================== layer 1 =====================
#pragma unroll
    for (int g = 0; g < 4; ++g) {
      f32x8 a;
#pragma unroll
      for (int v = 0; v < 8; ++v) a[v] = bias1[g];
      acc[g] = a;
    }
    gemm_k64_split(&lds_w[1 * 32768 + wv4 * 8192],   // h0_t @ Wih1^T
                   &lds_hb[((0 * 2 + (1 - pb)) * 2 + 0) * HB_PLANE + arow * HB_RS + aoff],
                   &lds_hb[((0 * 2 + (1 - pb)) * 2 + 1) * HB_PLANE + arow * HB_RS + aoff],
                   lane, acc);
    gemm_k64_split(&lds_w[2 * 32768 + wv4 * 8192],   // h1_{t-1} @ Whh1^T
                   &lds_hb[((1 * 2 + pb) * 2 + 0) * HB_PLANE + arow * HB_RS + aoff],
                   &lds_hb[((1 * 2 + pb) * 2 + 1) * HB_PLANE + arow * HB_RS + aoff],
                   lane, acc);
    cell_update(acc, c1,
                &lds_hb[((1 * 2 + (1 - pb)) * 2 + 0) * HB_PLANE],
                &lds_hb[((1 * 2 + (1 - pb)) * 2 + 1) * HB_PLANE], mt, lane, wv4);
    __syncthreads();
    pb ^= 1;
  }

  // ---- final FC: out = (hi+lo) @ Wfc^T + bfc ----
  if (tid < MTILE * OUT_F) {
    int r = tid & 31;
    int o = tid >> 5;
    const __bf16* hhi = &lds_hb[((1 * 2 + pb) * 2 + 0) * HB_PLANE + r * HB_RS];
    const __bf16* hlo = hhi + HB_PLANE;
    float s = bfc[o];
#pragma unroll
    for (int k = 0; k < H; ++k)
      s += ((float)hhi[k] + (float)hlo[k]) * Wfc[o * H + k];
    out[(size_t)(b0 + r) * OUT_F + o] = s;
  }
}

extern "C" void kernel_launch(void* const* d_in, const int* in_sizes, int n_in,
                              void* d_out, int out_size, void* d_ws, size_t ws_size,
                              hipStream_t stream) {
  (void)in_sizes; (void)n_in; (void)d_ws; (void)ws_size; (void)out_size;
  const float* x    = (const float*)d_in[0];
  const float* Wih0 = (const float*)d_in[1];
  const float* Whh0 = (const float*)d_in[2];
  const float* bih0 = (const float*)d_in[3];
  const float* bhh0 = (const float*)d_in[4];
  const float* Wih1 = (const float*)d_in[5];
  const float* Whh1 = (const float*)d_in[6];
  const float* bih1 = (const float*)d_in[7];
  const float* bhh1 = (const float*)d_in[8];
  const float* Wfc  = (const float*)d_in[9];
  const float* bfc  = (const float*)d_in[10];

  dim3 grid(4096 / MTILE);   // 128 WGs
  dim3 block(THREADS);
  lstm2_fused_wmma<<<grid, block, 0, stream>>>(x, Wih0, Whh0, bih0, bhh0,
                                               Wih1, Whh1, bih1, bhh1,
                                               Wfc, bfc, (float*)d_out);
}
